// Net_59021440582147
// MI455X (gfx1250) — compile-verified, matrix operand order checked by reading, not run on hardware
//
#include <hip/hip_runtime.h>
#include <hip/hip_bf16.h>

// ---------------------------------------------------------------------------
// GraphSAGE 2-layer pipeline for MI455X (gfx1250, wave32, WMMA).
// bf16 WMMA (v_wmma_f32_16x16x32_bf16) GEMMs with fp32 accumulation;
// async global->LDS staging (GLOBAL_LOAD_ASYNC_TO_LDS_B128, ASYNCcnt);
// fp32 global atomics for mean-aggregation; bf16 feature storage to halve
// HBM traffic on the ~7.7GB aggregation gather.
// ---------------------------------------------------------------------------

typedef __bf16 bf16;
typedef __attribute__((ext_vector_type(16))) __bf16 v16bf;
typedef __attribute__((ext_vector_type(8)))  float  v8f;
typedef __attribute__((ext_vector_type(4)))  unsigned int u32x4;

#define N_NODES 20000
#define N_EDGES 640000
#define DIM_IN  1280
#define DIM_HID 6000
#define NUM_CLS 5000
#define BN_EPS  1e-5f

// ---- CDNA5 async global->LDS copy (ISA 15.18.3 op 98, ASYNCcnt) ----------
__device__ __forceinline__ void async_load_b128(unsigned lds_byte_addr,
                                                const void* gaddr) {
  asm volatile("global_load_async_to_lds_b128 %0, %1, off"
               :
               : "v"(lds_byte_addr), "v"(gaddr)
               : "memory");
}

__device__ __forceinline__ void wait_async0() {
#if defined(__has_builtin)
#if __has_builtin(__builtin_amdgcn_s_wait_asynccnt)
  __builtin_amdgcn_s_wait_asynccnt(0);
  return;
#else
  asm volatile("s_wait_asynccnt 0x0" ::: "memory");
  return;
#endif
#else
  asm volatile("s_wait_asynccnt 0x0" ::: "memory");
#endif
}

// ======================= utility kernels ==================================

__global__ void k_zero_f32(float* __restrict__ p, long long n) {
  long long i = blockIdx.x * (long long)blockDim.x + threadIdx.x;
  if (i < n) p[i] = 0.f;
}

__global__ void k_cast_bf16(const float* __restrict__ s, bf16* __restrict__ d,
                            long long n) {
  long long i = blockIdx.x * (long long)blockDim.x + threadIdx.x;
  if (i < n) d[i] = (bf16)s[i];
}

__global__ void k_degree(const int* __restrict__ dst, float* __restrict__ deg,
                         int E) {
  int e = blockIdx.x * blockDim.x + threadIdx.x;
  if (e < E) atomicAdd(&deg[dst[e]], 1.f);
}

// Gather feat[src[e]] and atomically accumulate into acc[dst[e]].
// 4 features per thread; bf16 gather (halves read traffic), fp32 accumulate.
__global__ void k_scatter4(const bf16* __restrict__ feat,
                           const int* __restrict__ src,
                           const int* __restrict__ dst,
                           float* __restrict__ acc, int E, int D) {
  long long gid = blockIdx.x * (long long)blockDim.x + threadIdx.x;
  int dq = D >> 2;
  long long total = (long long)E * dq;
  if (gid >= total) return;
  int e = (int)(gid / dq);
  int f = (int)(gid % dq) * 4;
  int s = src[e], d = dst[e];
  const bf16* p = feat + (size_t)s * D + f;
  float* q = acc + (size_t)d * D + f;
  atomicAdd(q + 0, (float)p[0]);
  atomicAdd(q + 1, (float)p[1]);
  atomicAdd(q + 2, (float)p[2]);
  atomicAdd(q + 3, (float)p[3]);
}

__global__ void k_finalize_mean(const float* __restrict__ acc,
                                const float* __restrict__ deg,
                                bf16* __restrict__ out, long long n, int D) {
  long long i = blockIdx.x * (long long)blockDim.x + threadIdx.x;
  if (i >= n) return;
  long long row = i / D;
  out[i] = (bf16)(acc[i] / fmaxf(deg[row], 1.f));
}

// Column-wise sum / sum-of-squares partial reduction (grid.y = row chunks).
__global__ void k_bn_partial(const bf16* __restrict__ x, float* __restrict__ s,
                             float* __restrict__ s2, int n, int d,
                             int rows_per) {
  int c = blockIdx.x * blockDim.x + threadIdx.x;
  if (c >= d) return;
  int r0 = blockIdx.y * rows_per;
  int r1 = r0 + rows_per; if (r1 > n) r1 = n;
  float a = 0.f, b = 0.f;
  for (int r = r0; r < r1; ++r) {
    float v = (float)x[(size_t)r * d + c];
    a += v; b += v * v;
  }
  atomicAdd(&s[c], a);
  atomicAdd(&s2[c], b);
}

__global__ void k_bn_finalize(const float* __restrict__ s,
                              const float* __restrict__ s2,
                              float* __restrict__ mu, float* __restrict__ rstd,
                              int n, int d) {
  int c = blockIdx.x * blockDim.x + threadIdx.x;
  if (c >= d) return;
  float m = s[c] / n;
  float v = s2[c] / n - m * m;
  mu[c] = m;
  rstd[c] = rsqrtf(v + BN_EPS);
}

__global__ void k_bn_norm_relu(bf16* __restrict__ x,
                               const float* __restrict__ mu,
                               const float* __restrict__ rstd,
                               const float* __restrict__ gamma,
                               const float* __restrict__ beta, int d) {
  int c = blockIdx.x * blockDim.x + threadIdx.x;
  int r = blockIdx.y;
  if (c >= d) return;
  size_t i = (size_t)r * d + c;
  float v = ((float)x[i] - mu[c]) * rstd[c] * gamma[c] + beta[c];
  x[i] = (bf16)fmaxf(v, 0.f);
}

// ======================= WMMA GEMM ========================================
// C[M,Nn] = act( A1[M,K1] * B1[Nn,K1]^T  (+ A2[M,K2] * B2[Nn,K2]^T) + bias )
// Block tile 128x128, BK=64 (16 WMMAs per barrier). 8 waves of 32 lanes,
// each wave owns 64x32 (4x2 accumulators, 2 K-steps). Global->LDS staging
// via global_load_async_to_lds_b128 (ASYNCcnt); OOB chunks zero-filled with
// ds_store_b128. LDS row stride 72 bf16 (144 B): 16B aligned, conflict-free
// (36*m mod 64 distinct for m=0..15). Requires K % 16 == 0.

#define BM 128
#define BN 128
#define BK 64
#define LDK 72
#define TILE_E (BM * LDK)  // 9216 bf16 = 18432 B per tile

__global__ __launch_bounds__(256)
void k_wmma_gemm_dual(const bf16* __restrict__ A1, const bf16* __restrict__ B1,
                      int K1, const bf16* __restrict__ A2,
                      const bf16* __restrict__ B2, int K2,
                      const float* __restrict__ bias, float* __restrict__ outF,
                      bf16* __restrict__ outB, int M, int Nn, int do_relu) {
  __shared__ __align__(16) bf16 smem[4 * TILE_E];  // {A,B} x double buffer

  const int tid = threadIdx.x;
  const int lane = tid & 31;
  const int wid = tid >> 5;
  const int wm = wid >> 2;   // 0..1 : wave row
  const int wn = wid & 3;    // 0..3 : wave col
  const int g  = lane >> 4;  // half-wave group
  const int ln = lane & 15;
  const int bm0 = blockIdx.x * BM;
  const int bn0 = blockIdx.y * BN;

  const int nt1 = (K1 + BK - 1) / BK;
  const int nt2 = (K2 > 0) ? (K2 + BK - 1) / BK : 0;
  const int nt = nt1 + nt2;

  v8f acc[4][2] = {};

  auto load_tiles = [&](int t, int buf) {
    const bf16* A; const bf16* B; int K, k0;
    if (t < nt1) { A = A1; B = B1; K = K1; k0 = t * BK; }
    else         { A = A2; B = B2; K = K2; k0 = (t - nt1) * BK; }
    bf16* As = smem + buf * 2 * TILE_E;
    bf16* Bs = As + TILE_E;
    const int row   = tid >> 1;        // 128 rows, 2 threads/row
    const int cbase = (tid & 1) * 32;  // each thread: 32 elems = 4 x b128
    const int gra = bm0 + row;
    const int grb = bn0 + row;
    const u32x4 z = {0, 0, 0, 0};
#pragma unroll
    for (int c = 0; c < 32; c += 8) {
      const int col = cbase + c;                 // 8-element (16B) chunk
      const bool kin = (k0 + col + 8) <= K;      // K % 16 == 0 => 8-granular ok
      bf16* la = As + row * LDK + col;
      if (gra < M && kin)
        async_load_b128((unsigned)(size_t)la, A + (size_t)gra * K + k0 + col);
      else
        *(u32x4*)la = z;
      bf16* lb = Bs + row * LDK + col;
      if (grb < Nn && kin)
        async_load_b128((unsigned)(size_t)lb, B + (size_t)grb * K + k0 + col);
      else
        *(u32x4*)lb = z;
    }
  };

  auto compute = [&](int buf) {
    const bf16* As = smem + buf * 2 * TILE_E;
    const bf16* Bs = As + TILE_E;
    union Frag { v16bf v; u32x4 u[2]; };
#pragma unroll
    for (int ks = 0; ks < BK; ks += 32) {
      Frag fa[4], fb[2];
      // A fragment: lane holds row (wm*64 + i*16 + ln); K elements are two
      // contiguous 8-element runs at k = g*8 and k = 16 + g*8 (ISA 7.12.2).
#pragma unroll
      for (int i = 0; i < 4; ++i) {
        const bf16* p = As + (wm * 64 + i * 16 + ln) * LDK + ks;
        fa[i].u[0] = *(const u32x4*)(p + g * 8);
        fa[i].u[1] = *(const u32x4*)(p + 16 + g * 8);
      }
#pragma unroll
      for (int j = 0; j < 2; ++j) {
        const bf16* p = Bs + (wn * 32 + j * 16 + ln) * LDK + ks;
        fb[j].u[0] = *(const u32x4*)(p + g * 8);
        fb[j].u[1] = *(const u32x4*)(p + 16 + g * 8);
      }
#pragma unroll
      for (int i = 0; i < 4; ++i)
#pragma unroll
        for (int j = 0; j < 2; ++j)
          acc[i][j] = __builtin_amdgcn_wmma_f32_16x16x32_bf16(
              false, fa[i].v, false, fb[j].v, (short)0, acc[i][j], false, false);
    }
  };

  if (nt > 0) load_tiles(0, 0);
  for (int t = 0; t < nt; ++t) {
    wait_async0();        // our async fills of the tile we are about to read
    __syncthreads();      // + everyone's (incl. zero-fill ds stores)
    if (t + 1 < nt) load_tiles(t + 1, (t + 1) & 1);
    compute(t & 1);
  }

  // Epilogue: C/D layout — lane ln = column, VGPR r -> row r + 8*g.
#pragma unroll
  for (int i = 0; i < 4; ++i) {
#pragma unroll
    for (int j = 0; j < 2; ++j) {
      int gn = bn0 + wn * 32 + j * 16 + ln;
      if (gn >= Nn) continue;
      float bv = bias ? bias[gn] : 0.f;
#pragma unroll
      for (int r = 0; r < 8; ++r) {
        int gm = bm0 + wm * 64 + i * 16 + 8 * g + r;
        if (gm >= M) continue;
        float v = acc[i][j][r] + bv;
        if (do_relu) v = fmaxf(v, 0.f);
        size_t idx = (size_t)gm * Nn + gn;
        if (outF) outF[idx] = v;
        if (outB) outB[idx] = (bf16)v;
      }
    }
  }
}

// ======================= host-side launch =================================

static inline long long cdiv_ll(long long a, long long b) { return (a + b - 1) / b; }

extern "C" void kernel_launch(void* const* d_in, const int* in_sizes, int n_in,
                              void* d_out, int out_size, void* d_ws,
                              size_t ws_size, hipStream_t stream) {
  (void)in_sizes; (void)n_in; (void)out_size; (void)ws_size;

  const float* x    = (const float*)d_in[0];
  const int*   ei   = (const int*)d_in[1];
  const float* Wp   = (const float*)d_in[2];
  const float* bp   = (const float*)d_in[3];
  const float* Wl1  = (const float*)d_in[4];
  const float* bl1  = (const float*)d_in[5];
  const float* Wr1  = (const float*)d_in[6];
  const float* gam  = (const float*)d_in[7];
  const float* bet  = (const float*)d_in[8];
  const float* Wl2  = (const float*)d_in[9];
  const float* bl2  = (const float*)d_in[10];
  const float* Wr2  = (const float*)d_in[11];
  const int* src = ei;
  const int* dst = ei + N_EDGES;

  // workspace bump allocator (256B aligned)
  char* base = (char*)d_ws;
  size_t off = 0;
  auto alloc = [&](size_t bytes) -> void* {
    off = (off + 255) & ~(size_t)255;
    void* p = base + off;
    off += bytes;
    return p;
  };
  bf16* xb    = (bf16*)alloc((size_t)N_NODES * DIM_IN * 2);
  bf16* hb    = (bf16*)alloc((size_t)N_NODES * DIM_IN * 2);
  float* agg1f = (float*)alloc((size_t)N_NODES * DIM_IN * 4);
  bf16* agg1b = (bf16*)alloc((size_t)N_NODES * DIM_IN * 2);
  bf16* outb  = (bf16*)alloc((size_t)N_NODES * DIM_HID * 2);
  float* agg2f = (float*)alloc((size_t)N_NODES * DIM_HID * 4);
  bf16* agg2b = (bf16*)alloc((size_t)N_NODES * DIM_HID * 2);
  float* deg   = (float*)alloc((size_t)N_NODES * 4);
  float* bns   = (float*)alloc((size_t)DIM_HID * 4);
  float* bns2  = (float*)alloc((size_t)DIM_HID * 4);
  float* mu    = (float*)alloc((size_t)DIM_HID * 4);
  float* rstd  = (float*)alloc((size_t)DIM_HID * 4);
  bf16* Wpb  = (bf16*)alloc((size_t)DIM_IN * DIM_IN * 2);
  bf16* Wl1b = (bf16*)alloc((size_t)DIM_HID * DIM_IN * 2);
  bf16* Wr1b = (bf16*)alloc((size_t)DIM_HID * DIM_IN * 2);
  bf16* Wl2b = (bf16*)alloc((size_t)NUM_CLS * DIM_HID * 2);
  bf16* Wr2b = (bf16*)alloc((size_t)NUM_CLS * DIM_HID * 2);

  const int T = 256;
  auto gz = [&](long long n) { return dim3((unsigned)cdiv_ll(n, T)); };

  // 0) zero accumulators (fresh every call: deterministic)
  k_zero_f32<<<gz((long long)N_NODES * DIM_IN), T, 0, stream>>>(agg1f, (long long)N_NODES * DIM_IN);
  k_zero_f32<<<gz((long long)N_NODES * DIM_HID), T, 0, stream>>>(agg2f, (long long)N_NODES * DIM_HID);
  k_zero_f32<<<gz(N_NODES), T, 0, stream>>>(deg, N_NODES);
  k_zero_f32<<<gz(DIM_HID), T, 0, stream>>>(bns, DIM_HID);
  k_zero_f32<<<gz(DIM_HID), T, 0, stream>>>(bns2, DIM_HID);

  // 1) casts to bf16
  k_cast_bf16<<<gz((long long)N_NODES * DIM_IN), T, 0, stream>>>(x, xb, (long long)N_NODES * DIM_IN);
  k_cast_bf16<<<gz((long long)DIM_IN * DIM_IN), T, 0, stream>>>(Wp, Wpb, (long long)DIM_IN * DIM_IN);
  k_cast_bf16<<<gz((long long)DIM_HID * DIM_IN), T, 0, stream>>>(Wl1, Wl1b, (long long)DIM_HID * DIM_IN);
  k_cast_bf16<<<gz((long long)DIM_HID * DIM_IN), T, 0, stream>>>(Wr1, Wr1b, (long long)DIM_HID * DIM_IN);
  k_cast_bf16<<<gz((long long)NUM_CLS * DIM_HID), T, 0, stream>>>(Wl2, Wl2b, (long long)NUM_CLS * DIM_HID);
  k_cast_bf16<<<gz((long long)NUM_CLS * DIM_HID), T, 0, stream>>>(Wr2, Wr2b, (long long)NUM_CLS * DIM_HID);

  // 2) GEMM1: h = relu(x @ Wp^T + bp) -> bf16
  {
    dim3 grid((unsigned)cdiv_ll(N_NODES, BM), (unsigned)cdiv_ll(DIM_IN, BN));
    k_wmma_gemm_dual<<<grid, 256, 0, stream>>>(xb, Wpb, DIM_IN, nullptr, nullptr, 0,
                                               bp, nullptr, hb, N_NODES, DIM_IN, 1);
  }

  // 3) aggregation 1: mean over incoming edges
  k_degree<<<gz(N_EDGES), T, 0, stream>>>(dst, deg, N_EDGES);
  k_scatter4<<<gz((long long)N_EDGES * (DIM_IN / 4)), T, 0, stream>>>(hb, src, dst, agg1f, N_EDGES, DIM_IN);
  k_finalize_mean<<<gz((long long)N_NODES * DIM_IN), T, 0, stream>>>(agg1f, deg, agg1b, (long long)N_NODES * DIM_IN, DIM_IN);

  // 4) GEMM2 (dual-K): out = agg1 @ Wl1^T + bl1 + x @ Wr1^T -> bf16
  {
    dim3 grid((unsigned)cdiv_ll(N_NODES, BM), (unsigned)cdiv_ll(DIM_HID, BN));
    k_wmma_gemm_dual<<<grid, 256, 0, stream>>>(agg1b, Wl1b, DIM_IN, xb, Wr1b, DIM_IN,
                                               bl1, nullptr, outb, N_NODES, DIM_HID, 0);
  }

  // 5) BatchNorm (training-mode batch stats) + ReLU, in place on outb
  {
    dim3 gp((unsigned)cdiv_ll(DIM_HID, T), 64);
    int rows_per = (N_NODES + 63) / 64;
    k_bn_partial<<<gp, T, 0, stream>>>(outb, bns, bns2, N_NODES, DIM_HID, rows_per);
    k_bn_finalize<<<gz(DIM_HID), T, 0, stream>>>(bns, bns2, mu, rstd, N_NODES, DIM_HID);
    dim3 gn((unsigned)cdiv_ll(DIM_HID, T), N_NODES);
    k_bn_norm_relu<<<gn, T, 0, stream>>>(outb, mu, rstd, gam, bet, DIM_HID);
  }

  // 6) aggregation 2 (the HBM-bound pass: ~7.7GB bf16 gather)
  k_scatter4<<<gz((long long)N_EDGES * (DIM_HID / 4)), T, 0, stream>>>(outb, src, dst, agg2f, N_EDGES, DIM_HID);
  k_finalize_mean<<<gz((long long)N_NODES * DIM_HID), T, 0, stream>>>(agg2f, deg, agg2b, (long long)N_NODES * DIM_HID, DIM_HID);

  // 7) GEMM3 (dual-K): d_out = agg2 @ Wl2^T + bl2 + out @ Wr2^T -> fp32
  {
    dim3 grid((unsigned)cdiv_ll(N_NODES, BM), (unsigned)cdiv_ll(NUM_CLS, BN));
    k_wmma_gemm_dual<<<grid, 256, 0, stream>>>(agg2b, Wl2b, DIM_HID, outb, Wr2b, DIM_HID,
                                               bl2, (float*)d_out, nullptr, N_NODES, NUM_CLS, 0);
  }
}
